// Mini_pointgnn_v12_41214506172888
// MI455X (gfx1250) — matrix operand order repaired and checked by reading, not compiled, not run on hardware
//
#include <hip/hip_runtime.h>

typedef __attribute__((ext_vector_type(16))) _Float16 v16h;
typedef __attribute__((ext_vector_type(8)))  _Float16 v8h;
typedef __attribute__((ext_vector_type(8)))  float    v8f;

#define LDR 96          // row stride (halves) of per-wave LDS row tile
#define WPB 8           // waves per block
#define TPB 256
#define RPW 32          // rows per wave (2 x 16-row WMMA tiles)

// ---------------- device helpers ----------------

// A-fragment (16x32 f16, M x K) per CDNA5 ISA layout:
// lanes 0-15: M=lane, elems 0..7 = K kb..kb+7, elems 8..15 = K kb+16..kb+23, kb=0
// lanes 16-31: same M=lane-16, kb=8
__device__ __forceinline__ v16h make_afrag(const _Float16* rows, int lane, int c) {
  int m  = lane & 15;
  int kb = c * 32 + ((lane >> 4) << 3);
  const _Float16* p = rows + m * LDR + kb;
  v8h lo = *(const v8h*)(p);
  v8h hi = *(const v8h*)(p + 16);
  return __builtin_shufflevector(lo, hi, 0,1,2,3,4,5,6,7,8,9,10,11,12,13,14,15);
}

// B-fragment (32x16 f16, K x N): lane L holds col N=L&15, K = (L>=16?8:0)+(e&7)+(e>>3)*16.
// Pre-packed fragment-contiguous: 16 halves per lane, fully coalesced.
__device__ __forceinline__ v16h load_bfrag(const _Float16* wpack, int c, int nt, int lane) {
  return *(const v16h*)(wpack + ((((c << 2) + nt) * 32 + lane) << 4));
}

// Two-tile MLP layer: D{0,1} = relu(A{0,1}*W + b).  Each B fragment is loaded
// once and feeds two WMMAs (operand-reuse blocking halves weight traffic).
template<int NK>
__device__ __forceinline__ void mlp_layer2(const _Float16* rows, const _Float16* wpack,
                                           const float* bias, int lane,
                                           v8f acc0[4], v8f acc1[4]) {
  v16h a0[NK], a1[NK];
#pragma unroll
  for (int c = 0; c < NK; ++c) {
    a0[c] = make_afrag(rows, lane, c);
    a1[c] = make_afrag(rows + 16 * LDR, lane, c);
  }
#pragma unroll
  for (int nt = 0; nt < 4; ++nt) {
    v8f d0 = {}, d1 = {};
#pragma unroll
    for (int c = 0; c < NK; ++c) {
      v16h b = load_bfrag(wpack, c, nt, lane);
      d0 = __builtin_amdgcn_wmma_f32_16x16x32_f16(false, a0[c], false, b,
                                                  (short)0, d0, false, false);
      d1 = __builtin_amdgcn_wmma_f32_16x16x32_f16(false, a1[c], false, b,
                                                  (short)0, d1, false, false);
    }
    float bv = bias[nt * 16 + (lane & 15)];
#pragma unroll
    for (int r = 0; r < 8; ++r) {
      float v0 = d0[r] + bv; d0[r] = v0 > 0.f ? v0 : 0.f;
      float v1 = d1[r] + bv; d1[r] = v1 > 0.f ? v1 : 0.f;
    }
    acc0[nt] = d0; acc1[nt] = d1;
  }
}

// Write D fragments (row-striped across lanes) back to LDS rows as f16 (K-striped).
__device__ __forceinline__ void store_hidden(_Float16* rows, int lane, const v8f acc[4]) {
  int n  = lane & 15;
  int mb = (lane >> 4) << 3;
#pragma unroll
  for (int nt = 0; nt < 4; ++nt)
#pragma unroll
    for (int r = 0; r < 8; ++r)
      rows[(mb + r) * LDR + nt * 16 + n] = (_Float16)acc[nt][r];
}

// ---------------- utility kernels ----------------

__global__ void zero_f32_kernel(float* __restrict__ p, int n) {
  int i = blockIdx.x * blockDim.x + threadIdx.x;
  if (i < n) p[i] = 0.f;
}

// Repack f32 row-major weight [din x dout] into f16 B-fragment order, zero padded.
__global__ void repack_weights_kernel(const float* __restrict__ w, int din, int dout,
                                      int nK, _Float16* __restrict__ wpack) {
  int idx = blockIdx.x * blockDim.x + threadIdx.x;
  if (idx >= nK * 2048) return;
  int e = idx & 15, lane = (idx >> 4) & 31, nt = (idx >> 9) & 3, c = idx >> 11;
  int n = nt * 16 + (lane & 15);
  int k = c * 32 + ((lane >= 16) ? 8 : 0) + (e & 7) + ((e >> 3) << 4);
  float v = (k < din && n < dout) ? w[k * dout + n] : 0.f;
  wpack[idx] = (_Float16)v;
}

// ---------------- GNN edge pass: per-edge MLP + atomic scatter-max ----------------

__global__ __launch_bounds__(TPB) void gnn_edge_kernel(
    const float* __restrict__ centers, const float* __restrict__ feats,
    const int* __restrict__ edges, int E,
    const _Float16* __restrict__ wp1, const float* __restrict__ b1,
    const _Float16* __restrict__ wp2, const float* __restrict__ b2,
    float* __restrict__ agg) {
  __shared__ v8h rowsAllv[WPB][(RPW * LDR) / 8];
  __shared__ int dAll[WPB][RPW];
  int wave = threadIdx.x >> 5, lane = threadIdx.x & 31;
  _Float16* rows = (_Float16*)rowsAllv[wave];
  int base = (blockIdx.x * WPB + wave) * RPW;
  int r = lane >> 1, p = lane & 1;
#pragma unroll
  for (int t = 0; t < 2; ++t) {
    int rt = r + t * 16;
    int ei = base + rt; int eic = ei < E ? ei : (E - 1);
    int s = edges[2 * eic], d = edges[2 * eic + 1];
    _Float16* row = rows + rt * LDR;
    if (p == 0) {
      dAll[wave][rt] = (ei < E) ? d : -1;
      row[0] = (_Float16)(centers[3 * s + 0] - centers[3 * d + 0]);
      row[1] = (_Float16)(centers[3 * s + 1] - centers[3 * d + 1]);
      row[2] = (_Float16)(centers[3 * s + 2] - centers[3 * d + 2]);
      for (int k = 67; k < 82; ++k) row[k] = (_Float16)0.f;
    } else {
      for (int k = 82; k < 96; ++k) row[k] = (_Float16)0.f;
    }
    const float4* fr = (const float4*)(feats + s * 64 + p * 32);
#pragma unroll
    for (int j = 0; j < 8; ++j) {
      float4 v = fr[j];
      int o = 3 + p * 32 + j * 4;
      row[o + 0] = (_Float16)v.x; row[o + 1] = (_Float16)v.y;
      row[o + 2] = (_Float16)v.z; row[o + 3] = (_Float16)v.w;
    }
  }
  v8f acc[2][4];
  mlp_layer2<3>(rows, wp1, b1, lane, acc[0], acc[1]);
  store_hidden(rows, lane, acc[0]);
  store_hidden(rows + 16 * LDR, lane, acc[1]);
  mlp_layer2<2>(rows, wp2, b2, lane, acc[0], acc[1]);
  // post-ReLU values >= 0: float max == int max on bit patterns, agg pre-zeroed
  int n = lane & 15, mb = (lane >> 4) << 3;
#pragma unroll
  for (int t = 0; t < 2; ++t)
#pragma unroll
    for (int nt = 0; nt < 4; ++nt)
#pragma unroll
      for (int rr = 0; rr < 8; ++rr) {
        int di = dAll[wave][t * 16 + mb + rr];
        if (di >= 0)
          atomicMax((int*)(agg + di * 64 + nt * 16 + n),
                    __float_as_int(acc[t][nt][rr]));
      }
}

// ---------------- GNN node pass: MLP(agg) + feats (+extra) ----------------

__global__ __launch_bounds__(TPB) void gnn_node_kernel(
    const float* __restrict__ agg, const float* __restrict__ feats,
    const float* __restrict__ extra, int M,
    const _Float16* __restrict__ wp1, const float* __restrict__ b1,
    const _Float16* __restrict__ wp2, const float* __restrict__ b2,
    float* __restrict__ out) {
  __shared__ v8h rowsAllv[WPB][(RPW * LDR) / 8];
  int wave = threadIdx.x >> 5, lane = threadIdx.x & 31;
  _Float16* rows = (_Float16*)rowsAllv[wave];
  int base = (blockIdx.x * WPB + wave) * RPW;
  int r = lane >> 1, p = lane & 1;
#pragma unroll
  for (int t = 0; t < 2; ++t) {
    int rt = r + t * 16;
    int i = base + rt; int ic = i < M ? i : (M - 1);
    _Float16* row = rows + rt * LDR;
    const float4* fr = (const float4*)(agg + ic * 64 + p * 32);
#pragma unroll
    for (int j = 0; j < 8; ++j) {
      float4 v = fr[j];
      int o = p * 32 + j * 4;
      row[o + 0] = (_Float16)v.x; row[o + 1] = (_Float16)v.y;
      row[o + 2] = (_Float16)v.z; row[o + 3] = (_Float16)v.w;
    }
  }
  v8f acc[2][4];
  mlp_layer2<2>(rows, wp1, b1, lane, acc[0], acc[1]);
  store_hidden(rows, lane, acc[0]);
  store_hidden(rows + 16 * LDR, lane, acc[1]);
  mlp_layer2<2>(rows, wp2, b2, lane, acc[0], acc[1]);
  int n = lane & 15, mb = (lane >> 4) << 3;
#pragma unroll
  for (int t = 0; t < 2; ++t)
#pragma unroll
    for (int nt = 0; nt < 4; ++nt)
#pragma unroll
      for (int rr = 0; rr < 8; ++rr) {
        int ii = base + t * 16 + mb + rr;
        if (ii < M) {
          int col = nt * 16 + n;
          float o = acc[t][nt][rr] + feats[ii * 64 + col];
          if (extra) o += extra[ii * 64 + col];
          out[ii * 64 + col] = o;
        }
      }
}

// ---------------- point FFN: MLP([points - c1[lab], remission]) -> atomicAdd f1 ----------------

__global__ __launch_bounds__(TPB) void point_ffn_kernel(
    const float* __restrict__ points, const float* __restrict__ remission,
    const float* __restrict__ c1, const int* __restrict__ labels, int N,
    const _Float16* __restrict__ wp1, const float* __restrict__ b1,
    const _Float16* __restrict__ wp2, const float* __restrict__ b2,
    float* __restrict__ f1) {
  __shared__ v8h rowsAllv[WPB][(RPW * LDR) / 8];
  __shared__ int labAll[WPB][RPW];
  int wave = threadIdx.x >> 5, lane = threadIdx.x & 31;
  _Float16* rows = (_Float16*)rowsAllv[wave];
  int base = (blockIdx.x * WPB + wave) * RPW;
  int r = lane >> 1, p = lane & 1;
#pragma unroll
  for (int t = 0; t < 2; ++t) {
    int rt = r + t * 16;
    int i = base + rt; int ic = i < N ? i : (N - 1);
    int lab = labels[ic];
    _Float16* row = rows + rt * LDR;
    if (p == 0) {
      labAll[wave][rt] = (i < N) ? lab : -1;
      row[0] = (_Float16)(points[3 * ic + 0] - c1[3 * lab + 0]);
      row[1] = (_Float16)(points[3 * ic + 1] - c1[3 * lab + 1]);
      row[2] = (_Float16)(points[3 * ic + 2] - c1[3 * lab + 2]);
      row[3] = (_Float16)remission[ic];
      for (int k = 4; k < 18; ++k) row[k] = (_Float16)0.f;
    } else {
      for (int k = 18; k < 32; ++k) row[k] = (_Float16)0.f;
    }
  }
  v8f acc[2][4];
  mlp_layer2<1>(rows, wp1, b1, lane, acc[0], acc[1]);
  store_hidden(rows, lane, acc[0]);
  store_hidden(rows + 16 * LDR, lane, acc[1]);
  mlp_layer2<2>(rows, wp2, b2, lane, acc[0], acc[1]);
  int n = lane & 15, mb = (lane >> 4) << 3;
#pragma unroll
  for (int t = 0; t < 2; ++t)
#pragma unroll
    for (int nt = 0; nt < 4; ++nt)
#pragma unroll
      for (int rr = 0; rr < 8; ++rr) {
        int li = labAll[wave][t * 16 + mb + rr];
        if (li >= 0) atomicAdd(f1 + li * 64 + nt * 16 + n, acc[t][nt][rr]);
      }
}

// ---------------- concat([cA[i]-cB[lab], feats[...]]) -> MLP -> store / atomic-max ----------------

template<int FEAT_BY_LABEL, int ATOMIC_MAX_OUT>
__global__ __launch_bounds__(TPB) void concat_mlp_kernel(
    const float* __restrict__ cA, const float* __restrict__ cB,
    const int* __restrict__ labels, const float* __restrict__ feats, int M,
    const _Float16* __restrict__ wp1, const float* __restrict__ b1,
    const _Float16* __restrict__ wp2, const float* __restrict__ b2,
    float* __restrict__ out) {
  __shared__ v8h rowsAllv[WPB][(RPW * LDR) / 8];
  __shared__ int labAll[WPB][RPW];
  int wave = threadIdx.x >> 5, lane = threadIdx.x & 31;
  _Float16* rows = (_Float16*)rowsAllv[wave];
  int base = (blockIdx.x * WPB + wave) * RPW;
  int r = lane >> 1, p = lane & 1;
#pragma unroll
  for (int t = 0; t < 2; ++t) {
    int rt = r + t * 16;
    int i = base + rt; int ic = i < M ? i : (M - 1);
    int lab = labels[ic];
    _Float16* row = rows + rt * LDR;
    if (p == 0) {
      labAll[wave][rt] = (i < M) ? lab : -1;
      row[0] = (_Float16)(cA[3 * ic + 0] - cB[3 * lab + 0]);
      row[1] = (_Float16)(cA[3 * ic + 1] - cB[3 * lab + 1]);
      row[2] = (_Float16)(cA[3 * ic + 2] - cB[3 * lab + 2]);
      for (int k = 67; k < 82; ++k) row[k] = (_Float16)0.f;
    } else {
      for (int k = 82; k < 96; ++k) row[k] = (_Float16)0.f;
    }
    int fi = FEAT_BY_LABEL ? lab : ic;
    const float4* fr = (const float4*)(feats + fi * 64 + p * 32);
#pragma unroll
    for (int j = 0; j < 8; ++j) {
      float4 v = fr[j];
      int o = 3 + p * 32 + j * 4;
      row[o + 0] = (_Float16)v.x; row[o + 1] = (_Float16)v.y;
      row[o + 2] = (_Float16)v.z; row[o + 3] = (_Float16)v.w;
    }
  }
  v8f acc[2][4];
  mlp_layer2<3>(rows, wp1, b1, lane, acc[0], acc[1]);
  store_hidden(rows, lane, acc[0]);
  store_hidden(rows + 16 * LDR, lane, acc[1]);
  mlp_layer2<2>(rows, wp2, b2, lane, acc[0], acc[1]);
  int n = lane & 15, mb = (lane >> 4) << 3;
#pragma unroll
  for (int t = 0; t < 2; ++t)
#pragma unroll
    for (int nt = 0; nt < 4; ++nt)
#pragma unroll
      for (int rr = 0; rr < 8; ++rr) {
        int col = nt * 16 + n;
        if (ATOMIC_MAX_OUT) {
          int li = labAll[wave][t * 16 + mb + rr];
          if (li >= 0)
            atomicMax((int*)(out + li * 64 + col),
                      __float_as_int(acc[t][nt][rr]));
        } else {
          int ii = base + t * 16 + mb + rr;
          if (ii < M) out[ii * 64 + col] = acc[t][nt][rr];
        }
      }
}

// ---------------- final: fbn MLP + 64x20 classifier ----------------

__global__ __launch_bounds__(TPB) void final_cls_kernel(
    const float* __restrict__ points, const float* __restrict__ c1,
    const int* __restrict__ labels, const float* __restrict__ f6, int N,
    const _Float16* __restrict__ wp1, const float* __restrict__ b1,
    const _Float16* __restrict__ wp2, const float* __restrict__ b2,
    const _Float16* __restrict__ wcls, const float* __restrict__ clsb,
    float* __restrict__ out) {
  __shared__ v8h rowsAllv[WPB][(RPW * LDR) / 8];
  int wave = threadIdx.x >> 5, lane = threadIdx.x & 31;
  _Float16* rows = (_Float16*)rowsAllv[wave];
  int base = (blockIdx.x * WPB + wave) * RPW;
  int r = lane >> 1, p = lane & 1;
#pragma unroll
  for (int t = 0; t < 2; ++t) {
    int rt = r + t * 16;
    int i = base + rt; int ic = i < N ? i : (N - 1);
    int lab = labels[ic];
    _Float16* row = rows + rt * LDR;
    if (p == 0) {
      row[0] = (_Float16)(points[3 * ic + 0] - c1[3 * lab + 0]);
      row[1] = (_Float16)(points[3 * ic + 1] - c1[3 * lab + 1]);
      row[2] = (_Float16)(points[3 * ic + 2] - c1[3 * lab + 2]);
      for (int k = 67; k < 82; ++k) row[k] = (_Float16)0.f;
    } else {
      for (int k = 82; k < 96; ++k) row[k] = (_Float16)0.f;
    }
    const float4* fr = (const float4*)(f6 + lab * 64 + p * 32);
#pragma unroll
    for (int j = 0; j < 8; ++j) {
      float4 v = fr[j];
      int o = 3 + p * 32 + j * 4;
      row[o + 0] = (_Float16)v.x; row[o + 1] = (_Float16)v.y;
      row[o + 2] = (_Float16)v.z; row[o + 3] = (_Float16)v.w;
    }
  }
  v8f acc[2][4];
  mlp_layer2<3>(rows, wp1, b1, lane, acc[0], acc[1]);
  store_hidden(rows, lane, acc[0]);
  store_hidden(rows + 16 * LDR, lane, acc[1]);
  mlp_layer2<2>(rows, wp2, b2, lane, acc[0], acc[1]);
  store_hidden(rows, lane, acc[0]);                 // hidden2 back to LDS
  store_hidden(rows + 16 * LDR, lane, acc[1]);
  v16h h0c0 = make_afrag(rows, lane, 0);
  v16h h0c1 = make_afrag(rows, lane, 1);
  v16h h1c0 = make_afrag(rows + 16 * LDR, lane, 0);
  v16h h1c1 = make_afrag(rows + 16 * LDR, lane, 1);
  int n = lane & 15, mb = (lane >> 4) << 3;
#pragma unroll
  for (int nt = 0; nt < 2; ++nt) {
    v16h b0 = load_bfrag(wcls, 0, nt, lane);
    v16h b1f = load_bfrag(wcls, 1, nt, lane);
    v8f d0 = {}, d1 = {};
    d0 = __builtin_amdgcn_wmma_f32_16x16x32_f16(false, h0c0, false, b0,
                                                (short)0, d0, false, false);
    d1 = __builtin_amdgcn_wmma_f32_16x16x32_f16(false, h1c0, false, b0,
                                                (short)0, d1, false, false);
    d0 = __builtin_amdgcn_wmma_f32_16x16x32_f16(false, h0c1, false, b1f,
                                                (short)0, d0, false, false);
    d1 = __builtin_amdgcn_wmma_f32_16x16x32_f16(false, h1c1, false, b1f,
                                                (short)0, d1, false, false);
    int col = nt * 16 + n;
#pragma unroll
    for (int rr = 0; rr < 8; ++rr) {
      int i0 = base + mb + rr;
      int i1 = base + 16 + mb + rr;
      if (i0 < N && col < 20) out[i0 * 20 + col] = d0[rr] + clsb[col];
      if (i1 < N && col < 20) out[i1 * 20 + col] = d1[rr] + clsb[col];
    }
  }
}

// ---------------- host ----------------

struct Mlp { const float *b1, *b2, *w1, *w2; int din; };
struct Pack { _Float16 *p1, *p2; };

extern "C" void kernel_launch(void* const* d_in, const int* in_sizes, int n_in,
                              void* d_out, int out_size, void* d_ws, size_t ws_size,
                              hipStream_t stream) {
  (void)out_size; (void)ws_size;
  if (n_in < 82) return;
  // jax pytree flatten order (sorted dict keys)
  const float* c1     = (const float*)d_in[0];
  const int*   e1     = (const int*)  d_in[1];
  const int*   l1lab  = (const int*)  d_in[2];
  const float* c2     = (const float*)d_in[3];
  const int*   e2     = (const int*)  d_in[4];
  const int*   l2lab  = (const int*)  d_in[5];
  const float* cls_b  = (const float*)d_in[6];
  const float* cls_w  = (const float*)d_in[7];
  const float* points = (const float*)d_in[80];
  const float* remiss = (const float*)d_in[81];
  int M1 = in_sizes[0] / 3, E1 = in_sizes[1] / 2, N = in_sizes[2];
  int M2 = in_sizes[3] / 3, E2 = in_sizes[4] / 2;

  auto getMlp = [&](int b, int din) -> Mlp {
    return Mlp{(const float*)d_in[b], (const float*)d_in[b + 1],
               (const float*)d_in[b + 2], (const float*)d_in[b + 3], din};
  };
  Mlp fbn = getMlp(8, 67),  ffn = getMlp(12, 4);
  Mlp g21e = getMlp(16, 67), g21o = getMlp(20, 64);
  Mlp g22e = getMlp(24, 67), g22o = getMlp(28, 64);
  Mlp g2e  = getMlp(32, 67), g2o  = getMlp(36, 64);
  Mlp g4e  = getMlp(40, 67), g4o  = getMlp(44, 64);
  Mlp g61e = getMlp(48, 67), g61o = getMlp(52, 64);
  Mlp g62e = getMlp(56, 67), g62o = getMlp(60, 64);
  Mlp g6e  = getMlp(64, 67), g6o  = getMlp(68, 64);
  Mlp l2m  = getMlp(72, 67), m2l  = getMlp(76, 67);

  char* wsp = (char*)d_ws; size_t off = 0;
  auto carve = [&](size_t bytes) -> void* {
    void* p = wsp + off; off += (bytes + 255) & ~(size_t)255; return p;
  };
  auto packMlp = [&](const Mlp& m) -> Pack {
    int nk1 = (m.din + 31) / 32;
    Pack pk;
    pk.p1 = (_Float16*)carve((size_t)nk1 * 2048 * 2);
    pk.p2 = (_Float16*)carve((size_t)2 * 2048 * 2);
    repack_weights_kernel<<<(nk1 * 2048 + 255) / 256, 256, 0, stream>>>(m.w1, m.din, 64, nk1, pk.p1);
    repack_weights_kernel<<<(2 * 2048 + 255) / 256, 256, 0, stream>>>(m.w2, 64, 64, 2, pk.p2);
    return pk;
  };
  Pack Pfbn = packMlp(fbn), Pffn = packMlp(ffn);
  Pack Pg2e = packMlp(g2e), Pg2o = packMlp(g2o);
  Pack Pg21e = packMlp(g21e), Pg21o = packMlp(g21o);
  Pack Pg22e = packMlp(g22e), Pg22o = packMlp(g22o);
  Pack Pg4e = packMlp(g4e), Pg4o = packMlp(g4o);
  Pack Pg6e = packMlp(g6e), Pg6o = packMlp(g6o);
  Pack Pg61e = packMlp(g61e), Pg61o = packMlp(g61o);
  Pack Pg62e = packMlp(g62e), Pg62o = packMlp(g62o);
  Pack Pl2m = packMlp(l2m), Pm2l = packMlp(m2l);
  _Float16* pcls = (_Float16*)carve((size_t)2 * 2048 * 2);
  repack_weights_kernel<<<(2 * 2048 + 255) / 256, 256, 0, stream>>>(cls_w, 64, 20, 2, pcls);

  size_t m1b = (size_t)M1 * 64 * 4, m2b = (size_t)M2 * 64 * 4;
  float* f1   = (float*)carve(m1b);
  float* f2   = (float*)carve(m1b);
  float* f21  = (float*)carve(m1b);
  float* f22  = (float*)carve(m1b);
  float* f5   = (float*)carve(m1b);
  float* fA   = (float*)carve(m1b);
  float* fB   = (float*)carve(m1b);
  float* agg1 = (float*)carve(m1b);
  float* f3   = (float*)carve(m2b);
  float* f4   = (float*)carve(m2b);
  float* agg2 = (float*)carve(m2b);

  auto zero = [&](float* p, int n) {
    zero_f32_kernel<<<(n + 255) / 256, 256, 0, stream>>>(p, n);
  };
  auto rowBlocks = [](int rowsN) { return (rowsN + RPW * WPB - 1) / (RPW * WPB); };
  auto gnnLayer = [&](const float* centers, const float* feats, const int* edges,
                      int E, int M, const Mlp& me, Pack pe, const Mlp& mo, Pack po,
                      const float* extra, float* aggbuf, float* outbuf) {
    zero(aggbuf, M * 64);
    gnn_edge_kernel<<<rowBlocks(E), TPB, 0, stream>>>(
        centers, feats, edges, E, pe.p1, me.b1, pe.p2, me.b2, aggbuf);
    gnn_node_kernel<<<rowBlocks(M), TPB, 0, stream>>>(
        aggbuf, feats, extra, M, po.p1, mo.b1, po.p2, mo.b2, outbuf);
  };

  // f1 = segment_sum(mlp(x, ffn), l1_labels)
  zero(f1, M1 * 64);
  point_ffn_kernel<<<rowBlocks(N), TPB, 0, stream>>>(
      points, remiss, c1, l1lab, N, Pffn.p1, ffn.b1, Pffn.p2, ffn.b2, f1);
  // f2, f2_1, f2_2
  gnnLayer(c1, f1,  e1, E1, M1, g2e,  Pg2e,  g2o,  Pg2o,  nullptr, agg1, f2);
  gnnLayer(c1, f2,  e1, E1, M1, g21e, Pg21e, g21o, Pg21o, nullptr, agg1, f21);
  gnnLayer(c1, f21, e1, E1, M1, g22e, Pg22e, g22o, Pg22o, nullptr, agg1, f22);
  // f3 = scatter_max(mlp(xm, m2l), l2_labels)
  zero(f3, M2 * 64);
  concat_mlp_kernel<0, 1><<<rowBlocks(M1), TPB, 0, stream>>>(
      c1, c2, l2lab, f22, M1, Pm2l.p1, m2l.b1, Pm2l.p2, m2l.b2, f3);
  // f4 = gnn on l2
  gnnLayer(c2, f3, e2, E2, M2, g4e, Pg4e, g4o, Pg4o, nullptr, agg2, f4);
  // f5 = mlp(xl, l2m)
  concat_mlp_kernel<1, 0><<<rowBlocks(M1), TPB, 0, stream>>>(
      c1, c2, l2lab, f4, M1, Pl2m.p1, l2m.b1, Pl2m.p2, l2m.b2, f5);
  // f6 chain with residuals
  gnnLayer(c1, f5, e1, E1, M1, g6e,  Pg6e,  g6o,  Pg6o,  f22, agg1, fA);
  gnnLayer(c1, fA, e1, E1, M1, g61e, Pg61e, g61o, Pg61o, f21, agg1, fB);
  gnnLayer(c1, fB, e1, E1, M1, g62e, Pg62e, g62o, Pg62o, f2,  agg1, fA);
  // f7 @ cls_w + cls_b
  final_cls_kernel<<<rowBlocks(N), TPB, 0, stream>>>(
      points, c1, l1lab, fA, N, Pfbn.p1, fbn.b1, Pfbn.p2, fbn.b2, pcls, cls_b,
      (float*)d_out);
}
